// EGNN_68539088109878
// MI455X (gfx1250) — matrix operand order repaired and checked by reading, not compiled, not run on hardware
//
#include <hip/hip_runtime.h>

typedef __attribute__((ext_vector_type(16))) _Float16 v16h;
typedef __attribute__((ext_vector_type(8)))  float    v8f;

#define HIDN   128
#define CELLF  20.0f
#define NNODE  32768
#define NEDGE  524288
#define NLAYER 4
#define WV     8          // waves per block (wave32)
#define ETPW   8          // edge tiles per wave
#define K1     258
#define K1P    288

static __device__ __forceinline__ float siluf(float x){ return x / (1.0f + __expf(-x)); }
static __device__ __forceinline__ float pmodf(float v){ return v - CELLF*floorf(v*(1.0f/CELLF)); }
static __device__ __forceinline__ float micf(float d){ return d - CELLF*rintf(d*(1.0f/CELLF)); }

static __device__ __forceinline__ void wave_sync(){
  __builtin_amdgcn_wave_barrier();
#if __has_builtin(__builtin_amdgcn_s_wait_dscnt)
  __builtin_amdgcn_s_wait_dscnt(0);
#endif
  __builtin_amdgcn_wave_barrier();
}

// A fragment: 16x32 f16 tile, rows M in a 16-row LDS tile with leading dim ldk.
// lane l: m=l&15, g=l>>4; halves 0..7 -> K = kbase+g*8..; halves 8..15 -> K = kbase+16+g*8..
static __device__ __forceinline__ v16h load_a_frag(const _Float16* t, int ldk, int kbase, int m, int g){
  const _Float16* p = t + m*ldk + kbase + g*8;
  union { v16h v; float4 q[2]; } u;
  u.q[0] = *(const float4*)(p);
  u.q[1] = *(const float4*)(p + 16);
  return u.v;
}
// B fragment from transposed weights wT[ncol][ldk]: lane l: n=l&15; halves 0..7 -> K = kbase+g*8..; halves 8..15 -> K = kbase+16+g*8..
static __device__ __forceinline__ v16h load_b_frag(const _Float16* wT, int ldk, int kbase, int ncol, int g){
  const _Float16* p = wT + (size_t)ncol*ldk + kbase + g*8;
  union { v16h v; float4 q[2]; } u;
  u.q[0] = *(const float4*)(p);
  u.q[1] = *(const float4*)(p + 16);
  return u.v;
}
static __device__ __forceinline__ v8f wmma16(v16h a, v16h b, v8f c){
  return __builtin_amdgcn_wmma_f32_16x16x32_f16(false, a, false, b, (short)0, c, false, false);
}

// global f32 W[K][128] -> LDS f16 transposed wT[128][Kpad], zero-pad K..Kpad
static __device__ __forceinline__ void stage_wT(_Float16* dst, const float* src, int K, int Kpad, int tid, int nthr){
  for (int idx = tid; idx < HIDN*Kpad; idx += nthr){
    int n = idx / Kpad, k = idx - n*Kpad;
    float v = (k < K) ? src[(size_t)k*HIDN + n] : 0.0f;
    dst[idx] = (_Float16)v;
  }
}
static __device__ __forceinline__ void stage_vecf(float* dst, const float* src, int n, int tid, int nthr){
  for (int i = tid; i < n; i += nthr) dst[i] = src[i];
}
static __device__ __forceinline__ void fatomic_add(float* p, float v){
  unsafeAtomicAdd(p, v);   // global_atomic_add_f32
}

// ---------------- fused edge kernel: edge MLP + coord MLP + scatter ----------------
__global__ void __launch_bounds__(WV*32) egnn_edge_kernel(
    const _Float16* __restrict__ hf, const float* __restrict__ xg,
    const int* __restrict__ row, const int* __restrict__ col,
    const float* __restrict__ eattr,
    const float* __restrict__ ew1, const float* __restrict__ eb1,
    const float* __restrict__ ew2, const float* __restrict__ eb2,
    const float* __restrict__ cw1, const float* __restrict__ cb1,
    const float* __restrict__ cwo,
    float* __restrict__ ssum, float* __restrict__ agg)
{
  extern __shared__ char smem[];
  _Float16* w1T = (_Float16*)smem;                 // 128*288
  _Float16* w2T = w1T + HIDN*K1P;                  // 128*128
  _Float16* wcT = w2T + HIDN*HIDN;                 // 128*128
  _Float16* stg = wcT + HIDN*HIDN;                 // WV*(16*288 + 16*128 + 16*128)
  float* fb   = (float*)(stg + WV*(16*K1P + 32*HIDN));  // eb1|eb2|cb1|cwo = 512
  float* phis = fb + 512;                          // WV*16
  int*   rls  = (int*)(phis + WV*16);              // WV*16

  const int tid = threadIdx.x, nthr = blockDim.x;
  stage_wT(w1T, ew1, K1, K1P, tid, nthr);
  stage_wT(w2T, ew2, HIDN, HIDN, tid, nthr);
  stage_wT(wcT, cw1, HIDN, HIDN, tid, nthr);
  stage_vecf(fb,      eb1, 128, tid, nthr);
  stage_vecf(fb+128,  eb2, 128, tid, nthr);
  stage_vecf(fb+256,  cb1, 128, tid, nthr);
  stage_vecf(fb+384,  cwo, 128, tid, nthr);
  __syncthreads();

  const int wv = tid >> 5, lane = tid & 31;
  const int m = lane & 15, g = lane >> 4;
  _Float16* ein  = stg + wv*(16*K1P + 32*HIDN);    // 16 x 288
  _Float16* act1 = ein + 16*K1P;                   // 16 x 128
  _Float16* act2 = act1 + 16*HIDN;                 // 16 x 128
  float* phi = phis + wv*16;
  int*   rl  = rls + wv*16;

  const int tile0 = (blockIdx.x*WV + wv)*ETPW;
  for (int ti = 0; ti < ETPW; ++ti){
    const int e = (tile0 + ti)*16 + m;
    const int re = row[e], ce = col[e];
    // software prefetch the next tile's gather rows while this tile computes
    if (ti + 1 < ETPW){
      const int e2 = e + 16;
      const _Float16* pf = hf + (size_t)(g ? col[e2] : row[e2])*HIDN;
      __builtin_prefetch(pf, 0, 1);
      __builtin_prefetch(pf + 64, 0, 1);
    }
    // gather h[row]->ein[:,0:128] (g==0 lanes), h[col]->ein[:,128:256] (g==1 lanes)
    {
      const float4* s = (const float4*)(hf + (size_t)(g ? ce : re)*HIDN);
      float4* d = (float4*)(ein + m*K1P + g*HIDN);
      #pragma unroll
      for (int j=0;j<16;++j) d[j] = s[j];
    }
    float d0=0.f, d1=0.f, d2=0.f;
    if (g == 0){
      rl[m] = re;
      d0 = micf(pmodf(xg[re*3+0]) - pmodf(xg[ce*3+0]));
      d1 = micf(pmodf(xg[re*3+1]) - pmodf(xg[ce*3+1]));
      d2 = micf(pmodf(xg[re*3+2]) - pmodf(xg[ce*3+2]));
      const float radial = d0*d0 + d1*d1 + d2*d2;
      ein[m*K1P + 256] = (_Float16)radial;
      ein[m*K1P + 257] = (_Float16)eattr[e];
      int* z = (int*)(ein + m*K1P + 258);
      #pragma unroll
      for (int j=0;j<15;++j) z[j] = 0;      // zero pad K=258..287
    }
    wave_sync();

    // GEMM1: ein(16x288) @ ew1 -> silu -> act1   (A fragments held in VGPRs)
    {
      v16h af[9];
      #pragma unroll
      for (int kt = 0; kt < 9; ++kt) af[kt] = load_a_frag(ein, K1P, kt*32, m, g);
      #pragma unroll
      for (int nt = 0; nt < 8; ++nt){
        v8f c = {};
        #pragma unroll
        for (int kt = 0; kt < 9; ++kt)
          c = wmma16(af[kt], load_b_frag(w1T, K1P, kt*32, nt*16 + m, g), c);
        const int ncol = nt*16 + m;
        const float bb = fb[ncol];
        union { v8f v; float f[8]; } cu; cu.v = c;
        #pragma unroll
        for (int r=0;r<8;++r)
          act1[(r + 8*g)*HIDN + ncol] = (_Float16)siluf(cu.f[r] + bb);
      }
    }
    wave_sync();
    // GEMM2: act1 @ ew2 -> silu -> m ; scatter-add into agg[row]
    {
      v16h af[4];
      #pragma unroll
      for (int kt = 0; kt < 4; ++kt) af[kt] = load_a_frag(act1, HIDN, kt*32, m, g);
      #pragma unroll
      for (int nt = 0; nt < 8; ++nt){
        v8f c = {};
        #pragma unroll
        for (int kt = 0; kt < 4; ++kt)
          c = wmma16(af[kt], load_b_frag(w2T, HIDN, kt*32, nt*16 + m, g), c);
        const int ncol = nt*16 + m;
        const float bb = fb[128 + ncol];
        union { v8f v; float f[8]; } cu; cu.v = c;
        #pragma unroll
        for (int r=0;r<8;++r){
          const float v = siluf(cu.f[r] + bb);
          act2[(r + 8*g)*HIDN + ncol] = (_Float16)v;
          fatomic_add(agg + (size_t)rl[r + 8*g]*HIDN + ncol, v);
        }
      }
    }
    wave_sync();
    // GEMM3: m @ cw1 -> silu -> dot cwo -> phi (per-edge scalar)
    float pa[8] = {0,0,0,0,0,0,0,0};
    {
      v16h af[4];
      #pragma unroll
      for (int kt = 0; kt < 4; ++kt) af[kt] = load_a_frag(act2, HIDN, kt*32, m, g);
      #pragma unroll
      for (int nt = 0; nt < 8; ++nt){
        v8f c = {};
        #pragma unroll
        for (int kt = 0; kt < 4; ++kt)
          c = wmma16(af[kt], load_b_frag(wcT, HIDN, kt*32, nt*16 + m, g), c);
        const int ncol = nt*16 + m;
        const float bb = fb[256 + ncol], ww = fb[384 + ncol];
        union { v8f v; float f[8]; } cu; cu.v = c;
        #pragma unroll
        for (int r=0;r<8;++r) pa[r] += siluf(cu.f[r] + bb) * ww;
      }
    }
    #pragma unroll
    for (int r=0;r<8;++r){
      float v = pa[r];
      v += __shfl_xor(v, 1); v += __shfl_xor(v, 2);
      v += __shfl_xor(v, 4); v += __shfl_xor(v, 8);
      if (m == 0) phi[r + 8*g] = v;
    }
    wave_sync();
    if (g == 0){
      const float p = phi[m];
      const int rb = rl[m]*3;
      fatomic_add(ssum + rb + 0, fminf(fmaxf(d0*p, -100.f), 100.f));
      fatomic_add(ssum + rb + 1, fminf(fmaxf(d1*p, -100.f), 100.f));
      fatomic_add(ssum + rb + 2, fminf(fmaxf(d2*p, -100.f), 100.f));
    }
    wave_sync();
  }
}

// ---------------- node kernel: vel MLP, x update, node MLP residual ----------------
__global__ void __launch_bounds__(WV*32) egnn_node_kernel(
    float* __restrict__ h, _Float16* __restrict__ hf,
    const float* __restrict__ agg, const float* __restrict__ ssum,
    const float* __restrict__ cnt, float* __restrict__ xg,
    const float* __restrict__ acc,
    const float* __restrict__ nw1, const float* __restrict__ nb1,
    const float* __restrict__ nw2, const float* __restrict__ nb2,
    const float* __restrict__ vw1, const float* __restrict__ vb1,
    const float* __restrict__ vw2, const float* __restrict__ vb2,
    int use_acc)
{
  extern __shared__ char smem[];
  _Float16* n1T = (_Float16*)smem;                // 128*256
  _Float16* n2T = n1T + HIDN*256;                 // 128*128
  _Float16* v1T = n2T + HIDN*HIDN;                // 128*128
  _Float16* stg = v1T + HIDN*HIDN;                // WV*(16*256 + 16*128)
  float* fb  = (float*)(stg + WV*(16*256 + 16*HIDN)); // nb1|nb2|vb1|vw2|vb2 (pad 640)
  float* vls = fb + 640;                          // WV*16

  const int tid = threadIdx.x, nthr = blockDim.x;
  stage_wT(n1T, nw1, 256, 256, tid, nthr);
  stage_wT(n2T, nw2, HIDN, HIDN, tid, nthr);
  stage_wT(v1T, vw1, HIDN, HIDN, tid, nthr);
  stage_vecf(fb,      nb1, 128, tid, nthr);
  stage_vecf(fb+128,  nb2, 128, tid, nthr);
  stage_vecf(fb+256,  vb1, 128, tid, nthr);
  stage_vecf(fb+384,  vw2, 128, tid, nthr);
  if (tid == 0) fb[512] = vb2[0];
  __syncthreads();

  const int wv = tid >> 5, lane = tid & 31;
  const int m = lane & 15, g = lane >> 4;
  _Float16* hin = stg + wv*(16*256 + 16*HIDN);    // 16 x 256 = [h | agg]
  _Float16* act = hin + 16*256;                   // 16 x 128
  float* vel = vls + wv*16;

  const int n0 = (blockIdx.x*WV + wv)*16;
  {
    const int n = n0 + m;
    if (g == 0){
      const float4* s = (const float4*)(hf + (size_t)n*HIDN);
      float4* d = (float4*)(hin + m*256);
      #pragma unroll
      for (int j=0;j<16;++j) d[j] = s[j];
    } else {
      const float* s = agg + (size_t)n*HIDN;
      _Float16* d = hin + m*256 + HIDN;
      #pragma unroll
      for (int j=0;j<HIDN;++j) d[j] = (_Float16)s[j];
    }
  }
  wave_sync();

  // hin fragments shared between vel-GEMM (kt 0..3 == h part) and node GEMM1 (kt 0..7)
  v16h va[8];
  #pragma unroll
  for (int kt = 0; kt < 8; ++kt) va[kt] = load_a_frag(hin, 256, kt*32, m, g);

  // vel = silu(h@vw1+vb1)@vw2 + vb2
  float pa[8] = {0,0,0,0,0,0,0,0};
  #pragma unroll
  for (int nt = 0; nt < 8; ++nt){
    v8f c = {};
    #pragma unroll
    for (int kt = 0; kt < 4; ++kt)
      c = wmma16(va[kt], load_b_frag(v1T, HIDN, kt*32, nt*16 + m, g), c);
    const int ncol = nt*16 + m;
    const float bb = fb[256+ncol], ww = fb[384+ncol];
    union { v8f v; float f[8]; } cu; cu.v = c;
    #pragma unroll
    for (int r=0;r<8;++r) pa[r] += siluf(cu.f[r] + bb) * ww;
  }
  #pragma unroll
  for (int r=0;r<8;++r){
    float v = pa[r];
    v += __shfl_xor(v,1); v += __shfl_xor(v,2); v += __shfl_xor(v,4); v += __shfl_xor(v,8);
    if (m == 0) vel[r + 8*g] = v;
  }
  wave_sync();
  if (g == 0){
    const int n = n0 + m;
    const float vv = vel[m] + fb[512];
    const float ic = 1.0f / fmaxf(cnt[n], 1.0f);
    #pragma unroll
    for (int k=0;k<3;++k){
      float dsp = ssum[n*3+k]*ic;
      if (use_acc) dsp += vv*acc[n*3+k];
      xg[n*3+k] = pmodf(xg[n*3+k]) + dsp;
    }
  }
  // node MLP: h += silu([h|agg]@nw1+nb1)@nw2 + nb2
  #pragma unroll
  for (int nt = 0; nt < 8; ++nt){
    v8f c = {};
    #pragma unroll
    for (int kt = 0; kt < 8; ++kt)
      c = wmma16(va[kt], load_b_frag(n1T, 256, kt*32, nt*16 + m, g), c);
    const int ncol = nt*16 + m;
    const float bb = fb[ncol];
    union { v8f v; float f[8]; } cu; cu.v = c;
    #pragma unroll
    for (int r=0;r<8;++r)
      act[(r+8*g)*HIDN + ncol] = (_Float16)siluf(cu.f[r] + bb);
  }
  wave_sync();
  {
    v16h af[4];
    #pragma unroll
    for (int kt = 0; kt < 4; ++kt) af[kt] = load_a_frag(act, HIDN, kt*32, m, g);
    #pragma unroll
    for (int nt = 0; nt < 8; ++nt){
      v8f c = {};
      #pragma unroll
      for (int kt = 0; kt < 4; ++kt)
        c = wmma16(af[kt], load_b_frag(n2T, HIDN, kt*32, nt*16 + m, g), c);
      const int ncol = nt*16 + m;
      const float bb = fb[128+ncol];
      union { v8f v; float f[8]; } cu; cu.v = c;
      #pragma unroll
      for (int r=0;r<8;++r){
        const size_t idx = (size_t)(n0 + r + 8*g)*HIDN + ncol;
        const float hn = h[idx] + cu.f[r] + bb;
        h[idx] = hn;
        hf[idx] = (_Float16)hn;
      }
    }
  }
}

// ---------------- time-embedding kernel: Fourier features + MLP, init h/x ----------------
__global__ void __launch_bounds__(WV*32) egnn_time_kernel(
    const int* __restrict__ feat, const float* __restrict__ t,
    const float* __restrict__ pos, const float* __restrict__ femb,
    const float* __restrict__ tW,
    const float* __restrict__ w1, const float* __restrict__ b1,
    const float* __restrict__ w2, const float* __restrict__ b2,
    float* __restrict__ h, _Float16* __restrict__ hf, float* __restrict__ xg)
{
  extern __shared__ char smem[];
  _Float16* w1T = (_Float16*)smem;        // 128*128
  _Float16* w2T = w1T + HIDN*HIDN;        // 128*128
  _Float16* stg = w2T + HIDN*HIDN;        // WV*(16*128 + 16*128)
  float* fb = (float*)(stg + WV*32*HIDN); // b1|b2|tW = 320
  const int tid = threadIdx.x, nthr = blockDim.x;
  stage_wT(w1T, w1, HIDN, HIDN, tid, nthr);
  stage_wT(w2T, w2, HIDN, HIDN, tid, nthr);
  stage_vecf(fb,     b1, 128, tid, nthr);
  stage_vecf(fb+128, b2, 128, tid, nthr);
  stage_vecf(fb+256, tW, 64, tid, nthr);
  __syncthreads();

  const int wv = tid >> 5, lane = tid & 31;
  const int m = lane & 15, g = lane >> 4;
  _Float16* tmb = stg + wv*32*HIDN;       // 16 x 128
  _Float16* act = tmb + 16*HIDN;          // 16 x 128
  const int n0 = (blockIdx.x*WV + wv)*16;
  {
    const int n = n0 + m;
    const float tv = t[n] * 6.283185307179586f;
    _Float16* d = tmb + m*HIDN + g*64;
    for (int j=0;j<64;++j){
      const float ang = tv * fb[256+j];
      d[j] = (_Float16)(g ? __cosf(ang) : __sinf(ang));
    }
    if (g == 0){
      xg[n*3+0] = pos[n*3+0]; xg[n*3+1] = pos[n*3+1]; xg[n*3+2] = pos[n*3+2];
    }
  }
  wave_sync();
  {
    v16h af[4];
    #pragma unroll
    for (int kt = 0; kt < 4; ++kt) af[kt] = load_a_frag(tmb, HIDN, kt*32, m, g);
    #pragma unroll
    for (int nt = 0; nt < 8; ++nt){
      v8f c = {};
      #pragma unroll
      for (int kt = 0; kt < 4; ++kt)
        c = wmma16(af[kt], load_b_frag(w1T, HIDN, kt*32, nt*16 + m, g), c);
      const int ncol = nt*16 + m;
      const float bb = fb[ncol];
      union { v8f v; float f[8]; } cu; cu.v = c;
      #pragma unroll
      for (int r=0;r<8;++r)
        act[(r+8*g)*HIDN + ncol] = (_Float16)siluf(cu.f[r] + bb);
    }
  }
  wave_sync();
  {
    v16h af[4];
    #pragma unroll
    for (int kt = 0; kt < 4; ++kt) af[kt] = load_a_frag(act, HIDN, kt*32, m, g);
    #pragma unroll
    for (int nt = 0; nt < 8; ++nt){
      v8f c = {};
      #pragma unroll
      for (int kt = 0; kt < 4; ++kt)
        c = wmma16(af[kt], load_b_frag(w2T, HIDN, kt*32, nt*16 + m, g), c);
      const int ncol = nt*16 + m;
      const float bb = fb[128+ncol];
      union { v8f v; float f[8]; } cu; cu.v = c;
      #pragma unroll
      for (int r=0;r<8;++r){
        const int n = n0 + r + 8*g;
        const float hv = cu.f[r] + bb + femb[(size_t)feat[n]*HIDN + ncol];
        h[(size_t)n*HIDN + ncol] = hv;
        hf[(size_t)n*HIDN + ncol] = (_Float16)hv;
      }
    }
  }
}

// ---------------- small helpers ----------------
__global__ void zero_kernel(float* p, int n){
  int i = blockIdx.x*blockDim.x + threadIdx.x;
  if (i < n) p[i] = 0.0f;
}
__global__ void count_kernel(const int* __restrict__ row, float* __restrict__ cnt){
  int e = blockIdx.x*blockDim.x + threadIdx.x;
  if (e < NEDGE) fatomic_add(&cnt[row[e]], 1.0f);
}
__global__ void pack_kernel(const float* __restrict__ h, const float* __restrict__ x,
                            float* __restrict__ out){
  int i = blockIdx.x*blockDim.x + threadIdx.x;
  if (i < NNODE*(HIDN+3)){
    int n = i / (HIDN+3), j = i - n*(HIDN+3);
    out[i] = (j < HIDN) ? h[(size_t)n*HIDN + j] : x[n*3 + (j-HIDN)];
  }
}

extern "C" void kernel_launch(void* const* d_in, const int* in_sizes, int n_in,
                              void* d_out, int out_size, void* d_ws, size_t ws_size,
                              hipStream_t stream)
{
  (void)in_sizes; (void)n_in; (void)out_size; (void)ws_size;
  const int*   feat = (const int*)d_in[0];
  const float* t    = (const float*)d_in[1];
  const float* pos  = (const float*)d_in[2];
  const float* acc  = (const float*)d_in[3];
  const int*   eidx = (const int*)d_in[4];
  const float* eatt = (const float*)d_in[5];
  const float* femb = (const float*)d_in[6];
  const float* tW   = (const float*)d_in[7];
  const float* t_w1 = (const float*)d_in[8];
  const float* t_b1 = (const float*)d_in[9];
  const float* t_w2 = (const float*)d_in[10];
  const float* t_b2 = (const float*)d_in[11];
  const float* ew1  = (const float*)d_in[12];
  const float* eb1  = (const float*)d_in[13];
  const float* ew2  = (const float*)d_in[14];
  const float* eb2  = (const float*)d_in[15];
  const float* nw1  = (const float*)d_in[16];
  const float* nb1  = (const float*)d_in[17];
  const float* nw2  = (const float*)d_in[18];
  const float* nb2  = (const float*)d_in[19];
  const float* cw1  = (const float*)d_in[20];
  const float* cb1  = (const float*)d_in[21];
  const float* cwo  = (const float*)d_in[22];
  const float* vw1  = (const float*)d_in[23];
  const float* vb1  = (const float*)d_in[24];
  const float* vw2  = (const float*)d_in[25];
  const float* vb2  = (const float*)d_in[26];
  const int* row = eidx;
  const int* col = eidx + NEDGE;

  char* ws = (char*)d_ws;
  float*    h   = (float*)ws;      ws += (size_t)NNODE*HIDN*4;
  float*    agg = (float*)ws;      ws += (size_t)NNODE*HIDN*4;
  _Float16* hf  = (_Float16*)ws;   ws += (size_t)NNODE*HIDN*2;
  float*    x   = (float*)ws;      ws += (size_t)NNODE*3*4;
  float*    ssm = (float*)ws;      ws += (size_t)NNODE*3*4;
  float*    cnt = (float*)ws;      ws += (size_t)NNODE*4;

  const size_t lds_time = 2*((size_t)2*HIDN*HIDN + (size_t)WV*32*HIDN) + 4*320;
  const size_t lds_edge = 2*((size_t)HIDN*K1P + 2*(size_t)HIDN*HIDN
                             + (size_t)WV*(16*K1P + 32*HIDN))
                          + 4*(512 + WV*16) + 4*(WV*16);
  const size_t lds_node = 2*((size_t)HIDN*256 + 2*(size_t)HIDN*HIDN
                             + (size_t)WV*(16*256 + 16*HIDN))
                          + 4*(640 + WV*16);

  (void)hipFuncSetAttribute((const void*)egnn_time_kernel,
      hipFuncAttributeMaxDynamicSharedMemorySize, (int)lds_time);
  (void)hipFuncSetAttribute((const void*)egnn_edge_kernel,
      hipFuncAttributeMaxDynamicSharedMemorySize, (int)lds_edge);
  (void)hipFuncSetAttribute((const void*)egnn_node_kernel,
      hipFuncAttributeMaxDynamicSharedMemorySize, (int)lds_node);

  zero_kernel<<<(NNODE+255)/256, 256, 0, stream>>>(cnt, NNODE);
  count_kernel<<<(NEDGE+255)/256, 256, 0, stream>>>(row, cnt);
  egnn_time_kernel<<<NNODE/16/WV, WV*32, lds_time, stream>>>(
      feat, t, pos, femb, tW, t_w1, t_b1, t_w2, t_b2, h, hf, x);

  for (int i = 0; i < NLAYER; ++i){
    zero_kernel<<<(NNODE*HIDN+255)/256, 256, 0, stream>>>(agg, NNODE*HIDN);
    zero_kernel<<<(NNODE*3+255)/256, 256, 0, stream>>>(ssm, NNODE*3);
    egnn_edge_kernel<<<NEDGE/16/WV/ETPW, WV*32, lds_edge, stream>>>(
        hf, x, row, col, eatt,
        ew1 + (size_t)i*K1*HIDN,   eb1 + i*HIDN,
        ew2 + (size_t)i*HIDN*HIDN, eb2 + i*HIDN,
        cw1 + (size_t)i*HIDN*HIDN, cb1 + i*HIDN,
        cwo + i*HIDN, ssm, agg);
    egnn_node_kernel<<<NNODE/16/WV, WV*32, lds_node, stream>>>(
        h, hf, agg, ssm, cnt, x, acc,
        nw1 + (size_t)i*256*HIDN,  nb1 + i*HIDN,
        nw2 + (size_t)i*HIDN*HIDN, nb2 + i*HIDN,
        vw1 + (size_t)i*HIDN*HIDN, vb1 + i*HIDN,
        vw2 + i*HIDN, vb2 + i, (i==0) ? 1 : 0);
  }
  pack_kernel<<<(NNODE*(HIDN+3)+255)/256, 256, 0, stream>>>(h, x, (float*)d_out);
}
